// ContrastiveLoss_dot_38448547234377
// MI455X (gfx1250) — compile-verified, matrix operand order checked by reading, not run on hardware
//
#include <hip/hip_runtime.h>

// ---------------------------------------------------------------------------
// MISA contrastive loss for MI455X (gfx1250, wave32, WMMA, async LDS copies).
//   K0: stream-convert fp32 im/s -> split-bf16 (hi + lo residual) planes in ws.
//   K1: per (caption i, 4-image block) compute 60x144 dot block with
//       V_WMMA_F32_16X16X32_BF16 (3 wmma per K=32: hh, hl, lh). Tiles are
//       staged in double-buffered LDS via GLOBAL_LOAD_ASYNC_TO_LDS_B128 so the
//       async unit streams chunk k+1 while WMMAs consume chunk k. Fused
//       max-over-regions + masked mean -> scores.
//   K2: contrastive hinge reduction over the 128x128 score matrix.
// ---------------------------------------------------------------------------

typedef __attribute__((ext_vector_type(16))) __bf16 v16bf;
typedef __attribute__((ext_vector_type(8)))  float  v8f;
typedef __attribute__((ext_vector_type(4)))  int    v4i;
typedef unsigned short u16;

#define B_   128
#define R_   36
#define W_   60
#define D_   1024
#define JB   4            // images per workgroup
#define BROWS (JB * R_)   // 144 B-rows = 9 N tiles, no padding
#define MT   64           // padded word dim (4 x 16)
#define KC   64           // K chunk staged in LDS
#define LDT  72           // LDS tile row stride (bf16 elems) = 144B (16B mult)
#define LDC  148          // LDS C row stride in floats
#define NTHREADS 384      // 12 waves
#define MARGIN_ 0.2f

// per-buffer u16 offsets inside one LDS stage buffer
#define OFF_AHI 0
#define OFF_ALO (MT * LDT)                        // 4608
#define OFF_BHI (2 * MT * LDT)                    // 9216
#define OFF_BLO (2 * MT * LDT + BROWS * LDT)      // 19584
#define BUF_U16 (2 * MT * LDT + 2 * BROWS * LDT)  // 29952 u16 = 59904 B

#define S_ELEMS  (B_ * W_ * D_)   // 7864320
#define IM_ELEMS (B_ * R_ * D_)   // 4718592

#if defined(__has_builtin)
#if __has_builtin(__builtin_amdgcn_global_load_async_to_lds_b128) && \
    __has_builtin(__builtin_amdgcn_s_wait_asynccnt)
#define HAVE_ASYNC_LDS 1
#endif
#endif
#ifndef HAVE_ASYNC_LDS
#define HAVE_ASYNC_LDS 0
#endif

typedef __attribute__((address_space(1))) v4i* gv4i_ptr;
typedef __attribute__((address_space(3))) v4i* lv4i_ptr;

union FragBF { v16bf v; uint4 q[2]; };

__device__ __forceinline__ u16 f32_to_bf16_rne(float f) {
    unsigned u = __float_as_uint(f);
    u += 0x7fffu + ((u >> 16) & 1u);
    return (u16)(u >> 16);
}
__device__ __forceinline__ float bf16_to_f32(u16 h) {
    return __uint_as_float(((unsigned)h) << 16);
}

__device__ __forceinline__ void copy16_to_lds(const u16* gsrc, u16* ldst) {
#if HAVE_ASYNC_LDS
    __builtin_amdgcn_global_load_async_to_lds_b128(
        (gv4i_ptr)(void*)(u16*)gsrc, (lv4i_ptr)(void*)ldst, 0, 0);
#else
    *(uint4*)(void*)ldst = *(const uint4*)(const void*)gsrc;
#endif
}
__device__ __forceinline__ void wait_async_lds() {
#if HAVE_ASYNC_LDS
    __builtin_amdgcn_s_wait_asynccnt(0);
#endif
}

// ---- K0: fp32 -> split bf16 (hi, lo) planes --------------------------------
__global__ __launch_bounds__(256)
void split_bf16_kernel(const float* __restrict__ im, const float* __restrict__ s,
                       u16* __restrict__ s_hi, u16* __restrict__ s_lo,
                       u16* __restrict__ im_hi, u16* __restrict__ im_lo) {
    // float4 slots: s = 1966080, im = 1179648, total = 3145728 = 12288 * 256
    const unsigned idx = blockIdx.x * 256u + threadIdx.x;
    const float* src;
    u16 *dh, *dl;
    unsigned off4;
    if (idx < (S_ELEMS / 4)) { src = s;  dh = s_hi;  dl = s_lo;  off4 = idx; }
    else                     { src = im; dh = im_hi; dl = im_lo; off4 = idx - (S_ELEMS / 4); }
    const float4 f = *(const float4*)(src + (size_t)off4 * 4);
    const u16 h0 = f32_to_bf16_rne(f.x), h1 = f32_to_bf16_rne(f.y);
    const u16 h2 = f32_to_bf16_rne(f.z), h3 = f32_to_bf16_rne(f.w);
    const u16 l0 = f32_to_bf16_rne(f.x - bf16_to_f32(h0));
    const u16 l1 = f32_to_bf16_rne(f.y - bf16_to_f32(h1));
    const u16 l2 = f32_to_bf16_rne(f.z - bf16_to_f32(h2));
    const u16 l3 = f32_to_bf16_rne(f.w - bf16_to_f32(h3));
    uint2 ph, pl;
    ph.x = (unsigned)h0 | ((unsigned)h1 << 16); ph.y = (unsigned)h2 | ((unsigned)h3 << 16);
    pl.x = (unsigned)l0 | ((unsigned)l1 << 16); pl.y = (unsigned)l2 | ((unsigned)l3 << 16);
    *(uint2*)(dh + (size_t)off4 * 4) = ph;
    *(uint2*)(dl + (size_t)off4 * 4) = pl;
}

// ---- K1: WMMA GEMM (double-buffered async pipeline) + fused MISA -----------
__global__ __launch_bounds__(NTHREADS)
void misa_gemm_kernel(const u16* __restrict__ s_hi, const u16* __restrict__ s_lo,
                      const u16* __restrict__ im_hi, const u16* __restrict__ im_lo,
                      const int* __restrict__ s_l, float* __restrict__ scores) {
    // LDS: two 59904B stage buffers (Ahi|Alo|Bhi|Blo each). After the GEMM the
    // front of the space is reused for C (64 x 148 f32 = 37888B) and a 256-f32
    // reduction scratch at byte offset 38912.
    __shared__ __align__(16) unsigned char smem[2 * 2 * BUF_U16];
    u16* lds = (u16*)smem;                    // buffer b at lds + b*BUF_U16
    float* smemC = (float*)smem;
    float* smemR = (float*)(smem + 38912);

    const int jb   = blockIdx.x;   // image block (4 images)
    const int i    = blockIdx.y;   // caption
    const int tid  = threadIdx.x;
    const int lane = tid & 31;
    const int wv   = tid >> 5;     // 0..11
    const int mt   = wv & 3;       // word tile 0..3
    const int nt0  = wv >> 2;      // base region tile 0..2 (+3, +6)
    const int hl   = lane >> 4;
    const int ml   = lane & 15;

    const size_t arow_g = (size_t)(i * W_) * D_;          // caption base row
    const size_t brow_g = (size_t)(jb * BROWS) * D_;      // image-block base row

    // issue one chunk's async copies into stage buffer `b` (no wait)
    auto issue_loads = [&](u16* b, int kc) {
        // 16B slots: A planes 480 each, B planes 1152 each -> 3264 total
        for (int t = tid; t < 3264; t += NTHREADS) {
            const u16* gsrc;
            u16* ldst;
            if (t < 960) {
                const int pl = t >= 480;
                const int u  = pl ? t - 480 : t;
                const int row = u >> 3, q = u & 7;
                gsrc = (pl ? s_lo : s_hi) + arow_g + (size_t)row * D_ + kc + q * 8;
                ldst = b + (pl ? OFF_ALO : OFF_AHI) + row * LDT + q * 8;
            } else {
                const int v  = t - 960;
                const int pl = v >= 1152;
                const int u  = pl ? v - 1152 : v;
                const int row = u >> 3, q = u & 7;
                gsrc = (pl ? im_lo : im_hi) + brow_g + (size_t)row * D_ + kc + q * 8;
                ldst = b + (pl ? OFF_BLO : OFF_BHI) + row * LDT + q * 8;
            }
            copy16_to_lds(gsrc, ldst);
        }
    };

    // zero padded A rows 60..63 (both planes, both buffers) once
    if (tid < 288) {
        const int pl  = tid >= 144;
        const int u   = pl ? tid - 144 : tid;
        const int row = W_ + (u / 36);
        const int cu  = u % 36;                  // uint column
        const int off = (pl ? OFF_ALO : OFF_AHI) + row * LDT;
        *((unsigned*)(lds + off) + cu) = 0u;
        *((unsigned*)(lds + BUF_U16 + off) + cu) = 0u;
    }

    v8f acc[3] = {};
    const int arow = (mt * 16 + ml) * LDT;

    issue_loads(lds, 0);             // prologue: stage chunk 0 into buffer 0
    wait_async_lds();
    __syncthreads();

    for (int kc = 0; kc < D_; kc += KC) {
        const int cur = (kc >> 6) & 1;
        u16* cb = lds + cur * BUF_U16;
        if (kc + KC < D_)
            issue_loads(lds + (cur ^ 1) * BUF_U16, kc + KC);  // stream next chunk

#pragma unroll
        for (int k0 = 0; k0 < KC; k0 += 32) {
            const int ko = k0 + 8 * hl;     // per-lane K offset (16B aligned)
            FragBF aH, aL;
            aH.q[0] = *(const uint4*)(const void*)(cb + OFF_AHI + arow + ko);
            aH.q[1] = *(const uint4*)(const void*)(cb + OFF_AHI + arow + ko + 16);
            aL.q[0] = *(const uint4*)(const void*)(cb + OFF_ALO + arow + ko);
            aL.q[1] = *(const uint4*)(const void*)(cb + OFF_ALO + arow + ko + 16);
#pragma unroll
            for (int kk = 0; kk < 3; ++kk) {
                const int brow = ((nt0 + 3 * kk) * 16 + ml) * LDT;
                FragBF bH, bL;
                bH.q[0] = *(const uint4*)(const void*)(cb + OFF_BHI + brow + ko);
                bH.q[1] = *(const uint4*)(const void*)(cb + OFF_BHI + brow + ko + 16);
                bL.q[0] = *(const uint4*)(const void*)(cb + OFF_BLO + brow + ko);
                bL.q[1] = *(const uint4*)(const void*)(cb + OFF_BLO + brow + ko + 16);
                acc[kk] = __builtin_amdgcn_wmma_f32_16x16x32_bf16(
                    false, aH.v, false, bH.v, (short)0, acc[kk], false, false);
                acc[kk] = __builtin_amdgcn_wmma_f32_16x16x32_bf16(
                    false, aH.v, false, bL.v, (short)0, acc[kk], false, false);
                acc[kk] = __builtin_amdgcn_wmma_f32_16x16x32_bf16(
                    false, aL.v, false, bH.v, (short)0, acc[kk], false, false);
            }
        }

        wait_async_lds();    // next chunk landed (overlapped with the WMMAs)
        __syncthreads();     // everyone done reading cb; safe to reuse
    }

    // ---- epilogue: spill C and reduce --------------------------------------
#pragma unroll
    for (int kk = 0; kk < 3; ++kk) {
        const int col = (nt0 + 3 * kk) * 16 + ml;
#pragma unroll
        for (int v = 0; v < 8; ++v)
            smemC[(mt * 16 + v + 8 * hl) * LDC + col] = acc[kk][v];
    }
    __syncthreads();

    // MISA: per (word w, image jj) max over 36 regions; mask by caption length.
    const int len = s_l[i];
    if (tid < 256) {
        const int w  = tid & 63;
        const int jj = tid >> 6;       // 0..3
        float contrib = 0.f;
        if (w < W_ && w < len) {
            const float* rowp = smemC + w * LDC + jj * R_;
            float mx = rowp[0];
#pragma unroll
            for (int r = 1; r < R_; ++r) mx = fmaxf(mx, rowp[r]);
            contrib = mx;
        }
        smemR[tid] = contrib;
    }
    __syncthreads();
    if (tid < JB) {
        float sum = 0.f;
#pragma unroll
        for (int t = 0; t < 64; ++t) sum += smemR[tid * 64 + t];
        scores[i * B_ + jb * JB + tid] = sum / (float)len;
    }
}

// ---- K2: contrastive hinge reduction ---------------------------------------
// With S[i,j] = sims[caption i, image j], the scalar equals
//   sum_{i!=j} relu(M + S[i,j] - S[j,j]) + relu(M + S[i,j] - S[i,i]).
__global__ __launch_bounds__(256)
void contrastive_loss_kernel(const float* __restrict__ S, float* __restrict__ out) {
    __shared__ float diag[B_];
    __shared__ float red[256];
    const int t = threadIdx.x;
    if (t < B_) diag[t] = S[t * B_ + t];
    __syncthreads();
    float acc = 0.f;
    for (int idx = t; idx < B_ * B_; idx += 256) {
        const int ii = idx >> 7;
        const int jj = idx & (B_ - 1);
        if (ii != jj) {
            const float v = S[idx];
            acc += fmaxf(MARGIN_ + v - diag[jj], 0.f)
                 + fmaxf(MARGIN_ + v - diag[ii], 0.f);
        }
    }
    red[t] = acc;
    __syncthreads();
    for (int off = 128; off > 0; off >>= 1) {
        if (t < off) red[t] += red[t + off];
        __syncthreads();
    }
    if (t == 0) out[0] = red[0];
}

extern "C" void kernel_launch(void* const* d_in, const int* in_sizes, int n_in,
                              void* d_out, int out_size, void* d_ws, size_t ws_size,
                              hipStream_t stream) {
    const float* im  = (const float*)d_in[0];   // (128, 36, 1024) f32
    const float* s   = (const float*)d_in[1];   // (128, 60, 1024) f32
    const int*   s_l = (const int*)d_in[2];     // (128,) i32
    (void)in_sizes; (void)n_in; (void)out_size; (void)ws_size;

    // Workspace layout (bytes):
    //   [0        ..   65536) scores 128x128 f32
    //   [65536    .. +15.7MB) s_hi   | then s_lo | im_hi | im_lo
    char* ws = (char*)d_ws;
    float* scores = (float*)ws;
    u16* s_hi  = (u16*)(ws + 65536);
    u16* s_lo  = s_hi  + S_ELEMS;
    u16* im_hi = s_lo  + S_ELEMS;
    u16* im_lo = im_hi + IM_ELEMS;

    split_bf16_kernel<<<(S_ELEMS + IM_ELEMS) / (4 * 256), 256, 0, stream>>>(
        im, s, s_hi, s_lo, im_hi, im_lo);
    misa_gemm_kernel<<<dim3(B_ / JB, B_), NTHREADS, 0, stream>>>(
        s_hi, s_lo, im_hi, im_lo, s_l, scores);
    contrastive_loss_kernel<<<1, 256, 0, stream>>>(scores, (float*)d_out);
}
